// DecisionStep_26018911879369
// MI455X (gfx1250) — matrix operand order repaired and checked by reading, not compiled, not run on hardware
//
#include <hip/hip_runtime.h>
#include <hip/hip_bf16.h>
#include <math.h>

// ---------------------------------------------------------------------------
// TabNet decision step on MI455X (gfx1250), wave32 + WMMA bf16.
//   mask = sparsemax(BN(a_prev@Wa+ba)*prior)
//   x    = GLU(BN((x_o*mask)@W0+b0))
//   x    = (GLU(BN(x@W1+b1)) + x)*sqrt(1/2)   (twice, W1 then W2)
//   d_i = relu(x[:, :512]); a_i = x[:, 512:]; prior_next = prior*(1.3-mask)
// GEMMs: v_wmma_f32_16x16x32_bf16, 8 WMMAs per wave per K-step,
// double-buffered LDS (load->regs | WMMA | regs->LDS | barrier).
// Workspace: 8192*2048*4 (Z) + 2*8192*1024*4 (X ping-pong) + 16KB ~= 134.25MB.
// ---------------------------------------------------------------------------

typedef __bf16 v16bf __attribute__((ext_vector_type(16)));
typedef __bf16 v8bf  __attribute__((ext_vector_type(8)));
typedef float  v8f   __attribute__((ext_vector_type(8)));

static constexpr int   Bsz  = 8192;
static constexpr int   IND  = 2048;   // IN_DIM
static constexpr int   FEAT = 1024;
static constexpr int   HALFD= 512;
static constexpr int   N2   = 2048;   // 2*FEAT == IN_DIM here
static constexpr float EPSV = 1e-5f;
static constexpr float SQH  = 0.70710678118654752f;
static constexpr float GAM  = 1.3f;

// GEMM tiling: 256 threads (8 wave32 as 2x4), block tile 64x256, K-step 32.
// Wave tile 32x64 -> 2x4 accumulators of 16x16 -> 8 WMMAs per K-step.
static constexpr int TBM = 64;
static constexpr int TBN = 256;
static constexpr int TBK = 32;
static constexpr int PA  = 40;  // LDS A pitch (bf16): 80B rows, 16B-aligned groups
static constexpr int PB  = 40;  // LDS B^T pitch (bf16)

// ---------------------------------------------------------------------------
// Z[M,N] = A[M,K] (optionally * Ascale elementwise) @ W[K,N] + bias
// fp32 inputs converted to bf16 while staging; double-buffered LDS.
// ---------------------------------------------------------------------------
__global__ __launch_bounds__(256)
void gemm_bf16_wmma_kernel(const float* __restrict__ A,
                           const float* __restrict__ Ascale,  // nullable
                           const float* __restrict__ W,
                           const float* __restrict__ bias,
                           float* __restrict__ Z,
                           int M, int K, int N)
{
    __shared__ __bf16 ldsA[2][TBM * PA];    // 2 x  5 KB
    __shared__ __bf16 ldsBT[2][TBN * PB];   // 2 x 20 KB

    const int tid  = threadIdx.x;
    const int lane = tid & 31;
    const int wave = tid >> 5;           // 0..7
    const int wm   = (wave >> 2) * 32;   // wave row offset: 0 / 32
    const int wn   = (wave & 3) * 64;    // wave col offset: 0/64/128/192
    const int m0   = blockIdx.y * TBM;
    const int n0   = blockIdx.x * TBN;

    const int lm  = lane & 15;           // M (A) / N (B,C) lane index
    const int hiL = lane >> 4;           // half-wave select
    const int kbA = hiL << 3;            // A: kbase 0 or 8  (ISA 16-bit A layout)
    const int ksB = hiL << 3;            // B: kbase 0 or 8  (same k order as A)

    v8f acc[2][4];
#pragma unroll
    for (int i = 0; i < 2; ++i)
#pragma unroll
        for (int j = 0; j < 4; ++j)
#pragma unroll
            for (int e = 0; e < 8; ++e) acc[i][j][e] = 0.0f;

    const int ar = tid >> 2;             // A stage: row 0..63
    const int ac = (tid & 3) << 3;       // A stage: col 0/8/16/24

    // staging registers (global -> regs -> LDS, overlapping WMMA)
    float4 a0, a1;
    float  breg[4][8];                   // thread owns column n0+tid, 4 K-octets

    const int nsteps = K / TBK;

    // ---- load_regs(kk): global fp32 -> registers -------------------------
    auto load_regs = [&](int kk) {
        const float* gp = A + (size_t)(m0 + ar) * K + (kk + ac);
        a0 = *(const float4*)gp;
        a1 = *(const float4*)(gp + 4);
        if (Ascale) {  // fused x = x_o * mask for GEMM0
            const float* sp = Ascale + (size_t)(m0 + ar) * K + (kk + ac);
            float4 s0 = *(const float4*)sp;
            float4 s1 = *(const float4*)(sp + 4);
            a0.x *= s0.x; a0.y *= s0.y; a0.z *= s0.z; a0.w *= s0.w;
            a1.x *= s1.x; a1.y *= s1.y; a1.z *= s1.z; a1.w *= s1.w;
        }
        // B: 8 K-strided dword loads per octet, coalesced across the 256
        // threads (consecutive tid -> consecutive n).
        const float* wp = W + (size_t)kk * N + (n0 + tid);
#pragma unroll
        for (int j = 0; j < 4; ++j)
#pragma unroll
            for (int i = 0; i < 8; ++i)
                breg[j][i] = wp[(size_t)(j * 8 + i) * N];
    };

    // ---- store_regs(buf): registers -> bf16 LDS, one b128 per chunk ------
    auto store_regs = [&](int buf) {
        v8bf pa;
        pa[0] = (__bf16)a0.x; pa[1] = (__bf16)a0.y;
        pa[2] = (__bf16)a0.z; pa[3] = (__bf16)a0.w;
        pa[4] = (__bf16)a1.x; pa[5] = (__bf16)a1.y;
        pa[6] = (__bf16)a1.z; pa[7] = (__bf16)a1.w;
        *(v8bf*)&ldsA[buf][ar * PA + ac] = pa;
#pragma unroll
        for (int j = 0; j < 4; ++j) {
            v8bf pb;
#pragma unroll
            for (int i = 0; i < 8; ++i) pb[i] = (__bf16)breg[j][i];
            *(v8bf*)&ldsBT[buf][tid * PB + j * 8] = pb;   // B^T row (n, K-octet)
        }
    };

    load_regs(0);
    store_regs(0);
    __syncthreads();

    for (int s = 0; s < nsteps; ++s) {
        const int  cur     = s & 1;
        const bool hasNext = (s + 1) < nsteps;
        if (hasNext) load_regs((s + 1) * TBK);   // global loads fly under WMMA

        // ---- fragments per CDNA5 16-bit WMMA layouts ----
        v16bf aF[2], bF[4];
#pragma unroll
        for (int i = 0; i < 2; ++i) {
            const __bf16* p = &ldsA[cur][(wm + i * 16 + lm) * PA + kbA];
            v8bf lo = *(const v8bf*)(p);        // K = kb+0..7
            v8bf hi = *(const v8bf*)(p + 16);   // K = kb+16..23
            aF[i] = __builtin_shufflevector(lo, hi, 0, 1, 2, 3, 4, 5, 6, 7,
                                            8, 9, 10, 11, 12, 13, 14, 15);
        }
#pragma unroll
        for (int j = 0; j < 4; ++j) {
            const __bf16* p = &ldsBT[cur][(wn + j * 16 + lm) * PB + ksB];
            v8bf lo = *(const v8bf*)(p);        // K = ks+0..7
            v8bf hi = *(const v8bf*)(p + 16);   // K = ks+16..23
            bF[j] = __builtin_shufflevector(lo, hi, 0, 1, 2, 3, 4, 5, 6, 7,
                                            8, 9, 10, 11, 12, 13, 14, 15);
        }

#pragma unroll
        for (int i = 0; i < 2; ++i)
#pragma unroll
            for (int j = 0; j < 4; ++j)
                acc[i][j] = __builtin_amdgcn_wmma_f32_16x16x32_bf16(
                    false, aF[i], false, bF[j], (short)0, acc[i][j], false, false);

        if (hasNext) store_regs(cur ^ 1);
        __syncthreads();
    }

    // ---- epilogue: + bias, store fp32 Z (C/D layout: m = i*16 + hi*8 + p) ----
#pragma unroll
    for (int j = 0; j < 4; ++j) {
        const int   n  = n0 + wn + j * 16 + lm;
        const float bv = bias[n];
#pragma unroll
        for (int i = 0; i < 2; ++i) {
            const int mb = m0 + wm + i * 16 + (hiL << 3);
#pragma unroll
            for (int p = 0; p < 8; ++p)
                Z[(size_t)(mb + p) * N + n] = acc[i][j][p] + bv;
        }
    }
}

// ---------------------------------------------------------------------------
// Deterministic per-column BN stats: thread-per-column, coalesced row scans.
// ---------------------------------------------------------------------------
__global__ __launch_bounds__(256)
void bn_stats_kernel(const float* __restrict__ Z, float* __restrict__ mu,
                     float* __restrict__ rsig, int M, int N)
{
    const int n = blockIdx.x * blockDim.x + threadIdx.x;
    if (n >= N) return;
    float s = 0.f, q = 0.f;
    for (int m = 0; m < M; ++m) {
        float v = Z[(size_t)m * N + n];
        s += v; q += v * v;
    }
    const float mean = s / (float)M;
    const float var  = q / (float)M - mean * mean;   // biased var, as jnp.var
    mu[n]   = mean;
    rsig[n] = rsqrtf(var + EPSV);
}

// ---------------------------------------------------------------------------
// Per-row sparsemax via bisection on the simplex-projection threshold tau:
// find tau in [zmax-1, zmax] with sum(max(z - tau, 0)) == 1 (f monotone).
// ---------------------------------------------------------------------------
__global__ __launch_bounds__(256)
void sparsemax_mask_kernel(const float* __restrict__ Z,
                           const float* __restrict__ g,
                           const float* __restrict__ bt,
                           const float* __restrict__ mu,
                           const float* __restrict__ rsig,
                           const float* __restrict__ prior,
                           float* __restrict__ mask_out,
                           float* __restrict__ prior_next)
{
    __shared__ float zrow[IND];
    __shared__ float red[256];
    const int    tid  = threadIdx.x;
    const size_t base = (size_t)blockIdx.x * IND;

    float lmax = -3.4e38f;
    for (int c = tid; c < IND; c += 256) {
        const float h = (Z[base + c] - mu[c]) * rsig[c] * g[c] + bt[c];
        const float z = h * prior[base + c];
        zrow[c] = z;
        lmax = fmaxf(lmax, z);
    }
    red[tid] = lmax; __syncthreads();
    for (int s = 128; s > 0; s >>= 1) {
        if (tid < s) red[tid] = fmaxf(red[tid], red[tid + s]);
        __syncthreads();
    }
    const float zmax = red[0];
    __syncthreads();
    for (int c = tid; c < IND; c += 256) zrow[c] -= zmax;  // now max == 0
    __syncthreads();

    float lo = -1.0f, hi = 0.0f;                            // f(lo)>=1, f(hi)<=0
    for (int it = 0; it < 40; ++it) {
        const float mid = 0.5f * (lo + hi);
        float ls = 0.f;
        for (int c = tid; c < IND; c += 256) ls += fmaxf(zrow[c] - mid, 0.f);
        red[tid] = ls; __syncthreads();
        for (int s = 128; s > 0; s >>= 1) {
            if (tid < s) red[tid] += red[tid + s];
            __syncthreads();
        }
        const float f = red[0];
        __syncthreads();
        if (f > 1.0f) lo = mid; else hi = mid;              // uniform branch
    }
    const float tau = 0.5f * (lo + hi);

    for (int c = tid; c < IND; c += 256) {
        const float mval = fmaxf(zrow[c] - tau, 0.f);
        mask_out[base + c]   = mval;
        prior_next[base + c] = prior[base + c] * (GAM - mval);
    }
}

// ---------------------------------------------------------------------------
// GLU(+optional residual): out = BN(Z)[:, :F] * sigmoid(BN(Z)[:, F:]) [ +res ]*c
// ---------------------------------------------------------------------------
__global__ __launch_bounds__(256)
void glu_kernel(const float* __restrict__ Z, const float* __restrict__ mu,
                const float* __restrict__ rsig, const float* __restrict__ g,
                const float* __restrict__ bt, const float* __restrict__ res,
                float* __restrict__ out)
{
    const int i = blockIdx.x * 256 + threadIdx.x;
    if (i >= Bsz * FEAT) return;
    const int    m  = i / FEAT, n = i % FEAT, n2 = n + FEAT;
    const size_t zb = (size_t)m * N2;
    const float h1 = (Z[zb + n]  - mu[n])  * rsig[n]  * g[n]  + bt[n];
    const float h2 = (Z[zb + n2] - mu[n2]) * rsig[n2] * g[n2] + bt[n2];
    float v = h1 * (1.0f / (1.0f + expf(-h2)));
    if (res) v = (v + res[i]) * SQH;
    out[i] = v;
}

// ---------------------------------------------------------------------------
// Head split: d_i = relu(x[:, :512]); a_i = x[:, 512:]
// ---------------------------------------------------------------------------
__global__ __launch_bounds__(256)
void head_kernel(const float* __restrict__ X, float* __restrict__ d_i,
                 float* __restrict__ a_i)
{
    const int i = blockIdx.x * 256 + threadIdx.x;
    if (i >= Bsz * FEAT) return;
    const int m = i / FEAT, n = i % FEAT;
    const float v = X[i];
    if (n < HALFD) d_i[(size_t)m * HALFD + n]           = fmaxf(v, 0.f);
    else           a_i[(size_t)m * HALFD + (n - HALFD)] = v;
}

// ---------------------------------------------------------------------------
extern "C" void kernel_launch(void* const* d_in, const int* in_sizes, int n_in,
                              void* d_out, int out_size, void* d_ws, size_t ws_size,
                              hipStream_t stream)
{
    (void)in_sizes; (void)n_in; (void)out_size; (void)ws_size;

    const float* a_prev = (const float*)d_in[0];
    const float* prior  = (const float*)d_in[1];
    const float* x_o    = (const float*)d_in[2];
    const float* Wa  = (const float*)d_in[3];
    const float* ba  = (const float*)d_in[4];
    const float* ga  = (const float*)d_in[5];
    const float* bta = (const float*)d_in[6];
    const float* W0  = (const float*)d_in[7];
    const float* b0  = (const float*)d_in[8];
    const float* g0  = (const float*)d_in[9];
    const float* bt0 = (const float*)d_in[10];
    const float* W1  = (const float*)d_in[11];
    const float* b1  = (const float*)d_in[12];
    const float* g1  = (const float*)d_in[13];
    const float* bt1 = (const float*)d_in[14];
    const float* W2  = (const float*)d_in[15];
    const float* b2  = (const float*)d_in[16];
    const float* g2  = (const float*)d_in[17];
    const float* bt2 = (const float*)d_in[18];

    // d_out = concat(d_i[B,512], a_i[B,512], mask[B,2048], prior_next[B,2048])
    float* out    = (float*)d_out;
    float* d_i    = out;
    float* a_i    = out + (size_t)Bsz * HALFD;
    float* maskp  = out + (size_t)2 * Bsz * HALFD;
    float* priorn = maskp + (size_t)Bsz * IND;

    // workspace layout
    char*  w   = (char*)d_ws;
    float* Z   = (float*)w;                                            // B x 2048
    float* X1  = (float*)(w + (size_t)Bsz * N2 * 4);                   // B x 1024
    float* X2  = (float*)(w + (size_t)Bsz * N2 * 4 + (size_t)Bsz * FEAT * 4);
    float* mu  = (float*)(w + (size_t)Bsz * N2 * 4 + 2 * (size_t)Bsz * FEAT * 4);
    float* rsg = mu + N2;

    const dim3 blk(256);
    const dim3 gGrid(N2 / TBN, Bsz / TBM);       // (8, 128)
    const int  sGrid = (N2 + 255) / 256;         // stats
    const int  eGrid = (Bsz * FEAT + 255) / 256; // elementwise

    // attention: Za = a_prev @ Wa + ba ; BN ; sparsemax -> mask, prior_next
    gemm_bf16_wmma_kernel<<<gGrid, blk, 0, stream>>>(a_prev, nullptr, Wa, ba, Z, Bsz, HALFD, N2);
    bn_stats_kernel<<<sGrid, blk, 0, stream>>>(Z, mu, rsg, Bsz, N2);
    sparsemax_mask_kernel<<<Bsz, blk, 0, stream>>>(Z, ga, bta, mu, rsg, prior, maskp, priorn);

    // stage 0: (x_o*mask) @ W0 fused ; BN ; GLU -> X1
    gemm_bf16_wmma_kernel<<<gGrid, blk, 0, stream>>>(x_o, maskp, W0, b0, Z, Bsz, IND, N2);
    bn_stats_kernel<<<sGrid, blk, 0, stream>>>(Z, mu, rsg, Bsz, N2);
    glu_kernel<<<eGrid, blk, 0, stream>>>(Z, mu, rsg, g0, bt0, nullptr, X1);

    // stage 1: X2 = (GLU(BN(X1@W1+b1)) + X1)*sqrt(1/2)
    gemm_bf16_wmma_kernel<<<gGrid, blk, 0, stream>>>(X1, nullptr, W1, b1, Z, Bsz, FEAT, N2);
    bn_stats_kernel<<<sGrid, blk, 0, stream>>>(Z, mu, rsg, Bsz, N2);
    glu_kernel<<<eGrid, blk, 0, stream>>>(Z, mu, rsg, g1, bt1, X1, X2);

    // stage 2: X1 = (GLU(BN(X2@W2+b2)) + X2)*sqrt(1/2)
    gemm_bf16_wmma_kernel<<<gGrid, blk, 0, stream>>>(X2, nullptr, W2, b2, Z, Bsz, FEAT, N2);
    bn_stats_kernel<<<sGrid, blk, 0, stream>>>(Z, mu, rsg, Bsz, N2);
    glu_kernel<<<eGrid, blk, 0, stream>>>(Z, mu, rsg, g2, bt2, X2, X1);

    // head
    head_kernel<<<eGrid, blk, 0, stream>>>(X1, d_i, a_i);
}
